// Codebook_4234837753958
// MI455X (gfx1250) — matrix operand order repaired and checked by reading, not compiled, not run on hardware
//
#include <hip/hip_runtime.h>
#include <hip/hip_bf16.h>

typedef __attribute__((ext_vector_type(16))) __bf16 v16bf;
typedef __attribute__((ext_vector_type(8)))  float  v8f;
typedef __attribute__((ext_vector_type(4)))  unsigned int v4u;
typedef __attribute__((ext_vector_type(4)))  int v4i;

#define NTOK   32768      // B*T*H*W = 8*4*32*32
#define NCODE  2048
#define KDIM   256
#define THW    4096       // T*H*W
#define ZTOT   8388608    // B*C*T*H*W

// ---- output element offsets (flat concat in reference return order) ----
#define O_ST    0          // 8388608  embeddings_st (forward value == quant)
#define O_ENC   8388608    // 32768    encodings (as float)
#define O_LOSS  8421376    // 1
#define O_PERP  8421377    // 1
#define O_NEWE  8421378    // 524288
#define O_NEWN  8945666    // 2048
#define O_NEWZ  8947714    // 524288

#define AS1 __attribute__((address_space(1)))
#define AS3 __attribute__((address_space(3)))

#if __has_builtin(__builtin_amdgcn_global_load_async_to_lds_b128) && \
    __has_builtin(__builtin_amdgcn_s_wait_asynccnt)
#define HAVE_ASYNC_LDS 1
#else
#define HAVE_ASYNC_LDS 0
#endif

static __device__ __forceinline__ v8f wmma_bf16(v16bf a, v16bf b, v8f c) {
    return __builtin_amdgcn_wmma_f32_16x16x32_bf16(false, a, false, b,
                                                   (short)0, c, false, false);
}

// copy one code-tile (8KB hi + 8KB lo) global -> LDS; 32B per thread per half.
static __device__ __forceinline__ void stage_tile(const __bf16* __restrict__ Bh,
                                                  const __bf16* __restrict__ Bl,
                                                  __bf16* dstH, __bf16* dstL,
                                                  int tn, int tid) {
    const char* gh = (const char*)(Bh + (size_t)tn * 4096) + tid * 32;
    const char* gl = (const char*)(Bl + (size_t)tn * 4096) + tid * 32;
    char* lh = (char*)dstH + tid * 32;
    char* ll = (char*)dstL + tid * 32;
#if HAVE_ASYNC_LDS
    __builtin_amdgcn_global_load_async_to_lds_b128((AS1 v4i*)gh, (AS3 v4i*)lh, 0, 0);
    __builtin_amdgcn_global_load_async_to_lds_b128((AS1 v4i*)gh, (AS3 v4i*)lh, 16, 0);
    __builtin_amdgcn_global_load_async_to_lds_b128((AS1 v4i*)gl, (AS3 v4i*)ll, 0, 0);
    __builtin_amdgcn_global_load_async_to_lds_b128((AS1 v4i*)gl, (AS3 v4i*)ll, 16, 0);
#else
    ((v4u*)lh)[0] = ((const v4u*)gh)[0];
    ((v4u*)lh)[1] = ((const v4u*)gh)[1];
    ((v4u*)ll)[0] = ((const v4u*)gl)[0];
    ((v4u*)ll)[1] = ((const v4u*)gl)[1];
#endif
}

static __device__ __forceinline__ void wait_async_le4() {
#if HAVE_ASYNC_LDS
    __builtin_amdgcn_s_wait_asynccnt(4);
#endif
}
static __device__ __forceinline__ void wait_async_le0() {
#if HAVE_ASYNC_LDS
    __builtin_amdgcn_s_wait_asynccnt(0);
#endif
}

__global__ void vq_zero_kernel(float* __restrict__ p, int n) {
    int i = blockIdx.x * 256 + threadIdx.x;
    if (i < n) p[i] = 0.f;
}

// z [8,256,4,32,32] -> WMMA A-operand swizzle (bf16 hi/lo) + ||x||^2
__global__ void vq_prep_a_kernel(const float* __restrict__ z,
                                 __bf16* __restrict__ Ah, __bf16* __restrict__ Al,
                                 float* __restrict__ xnorm) {
    int gid   = blockIdx.x * 256 + threadIdx.x;   // 524288 threads
    int lane  = gid & 31;
    int kstep = (gid >> 5) & 7;
    int tileM = gid >> 8;
    int half  = lane >> 4;
    int token = tileM * 16 + (lane & 15);
    int b = token >> 12, thw = token & 4095;
    float ss = 0.f;
    __bf16 hv[16], lv[16];
#pragma unroll
    for (int j = 0; j < 16; ++j) {
        int v = j >> 1, e = j & 1;
        int kloc = ((v < 4) ? 0 : 16) + half * 8 + (v & 3) * 2 + e;  // A layout (ISA 7.12.2)
        int k = kstep * 32 + kloc;
        float x = z[((size_t)(b * KDIM + k)) * THW + thw];
        __bf16 h = (__bf16)x;
        hv[j] = h;
        lv[j] = (__bf16)(x - (float)h);
        ss += x * x;
    }
    size_t base = (size_t)gid * 16;
#pragma unroll
    for (int j = 0; j < 16; ++j) { Ah[base + j] = hv[j]; Al[base + j] = lv[j]; }
    atomicAdd(&xnorm[token], ss);
}

// embeddings [2048,256] -> WMMA B-operand swizzle (bf16 hi/lo) + ||e||^2
__global__ void vq_prep_b_kernel(const float* __restrict__ emb,
                                 __bf16* __restrict__ Bh, __bf16* __restrict__ Bl,
                                 float* __restrict__ enorm) {
    int gid   = blockIdx.x * 256 + threadIdx.x;   // 32768 threads
    int lane  = gid & 31;
    int kstep = (gid >> 5) & 7;
    int tileN = gid >> 8;
    int half  = lane >> 4;
    int n = tileN * 16 + (lane & 15);
    float ss = 0.f;
    __bf16 hv[16], lv[16];
#pragma unroll
    for (int j = 0; j < 16; ++j) {
        int kloc = ((j < 8) ? 0 : 16) + half * 8 + (j & 7);     // B layout: same K order as A
        int k = kstep * 32 + kloc;
        float x = emb[n * KDIM + k];
        __bf16 h = (__bf16)x;
        hv[j] = h;
        lv[j] = (__bf16)(x - (float)h);
        ss += x * x;
    }
    size_t base = (size_t)gid * 16;
#pragma unroll
    for (int j = 0; j < 16; ++j) { Bh[base + j] = hv[j]; Bl[base + j] = lv[j]; }
    atomicAdd(&enorm[n], ss);
}

// Block = 8 waves, each wave owns one 16-token tile; B tiles are shared,
// double-buffered in LDS via async global->LDS copies (ASYNCcnt pipeline).
__global__ void __launch_bounds__(256)
vq_argmin_kernel(const __bf16* __restrict__ Ah, const __bf16* __restrict__ Al,
                 const __bf16* __restrict__ Bh, const __bf16* __restrict__ Bl,
                 const float* __restrict__ xnorm, const float* __restrict__ enorm,
                 int* __restrict__ idx) {
    __shared__ __align__(32) __bf16 sB[2][2][4096];   // [buf][hi/lo][8KB]
    int tid  = threadIdx.x;
    int lane = tid & 31;
    int wave = blockIdx.x * 8 + (tid >> 5);           // tileM in [0,2048)

    // A operands: register-resident for the whole code sweep
    const v16bf* Aph = (const v16bf*)Ah + (size_t)wave * 8 * 32 + lane;
    const v16bf* Apl = (const v16bf*)Al + (size_t)wave * 8 * 32 + lane;
    v16bf a_hi[8], a_lo[8];
#pragma unroll
    for (int ks = 0; ks < 8; ++ks) { a_hi[ks] = Aph[ks * 32]; a_lo[ks] = Apl[ks * 32]; }

    int m0 = wave * 16 + (lane >> 4) * 8;   // D layout: half-wave holds rows m0..m0+7
    float xn[8];
#pragma unroll
    for (int r = 0; r < 8; ++r) xn[r] = xnorm[m0 + r];
    float best[8]; int bidx[8];
#pragma unroll
    for (int r = 0; r < 8; ++r) { best[r] = 3.4e38f; bidx[r] = 0; }

    // prologue: stage tile 0 into buffer 0
    stage_tile(Bh, Bl, &sB[0][0][0], &sB[0][1][0], 0, tid);

    for (int tn = 0; tn < NCODE / 16; ++tn) {
        int cur = tn & 1;
        if (tn + 1 < NCODE / 16) {
            // safe: everyone passed the tail barrier of iteration tn-1,
            // so nobody is still reading buf[cur^1]
            stage_tile(Bh, Bl, &sB[cur ^ 1][0][0], &sB[cur ^ 1][1][0], tn + 1, tid);
            wait_async_le4();   // async loads complete in order: tile tn has landed
        } else {
            wait_async_le0();
        }
        __syncthreads();        // all waves' copies for buf[cur] visible

        // preload full B tile into registers, then issue WMMAs back-to-back
        const v16bf* ph = (const v16bf*)(&sB[cur][0][0]) + lane;
        const v16bf* pl = (const v16bf*)(&sB[cur][1][0]) + lane;
        v16bf b_hi[8], b_lo[8];
#pragma unroll
        for (int ks = 0; ks < 8; ++ks) { b_hi[ks] = ph[ks * 32]; b_lo[ks] = pl[ks * 32]; }

        v8f acc = {0.f, 0.f, 0.f, 0.f, 0.f, 0.f, 0.f, 0.f};
#pragma unroll
        for (int ks = 0; ks < 8; ++ks) {
            acc = wmma_bf16(a_hi[ks], b_hi[ks], acc);   // hi*hi
            acc = wmma_bf16(a_lo[ks], b_hi[ks], acc);   // lo*hi
            acc = wmma_bf16(a_hi[ks], b_lo[ks], acc);   // hi*lo (~fp32 accuracy)
        }
        int n = tn * 16 + (lane & 15);
        float en = enorm[n];
#pragma unroll
        for (int r = 0; r < 8; ++r) {
            float d = fmaf(-2.f, acc[r], xn[r] + en);
            if (d < best[r]) { best[r] = d; bidx[r] = n; }
        }
        __syncthreads();        // done reading buf[cur]; it may be overwritten next iter
    }

    // butterfly min-reduce across the 16 lanes sharing each M row
#pragma unroll
    for (int off = 8; off >= 1; off >>= 1) {
#pragma unroll
        for (int r = 0; r < 8; ++r) {
            float ov = __shfl_xor(best[r], off, 16);
            int   oi = __shfl_xor(bidx[r], off, 16);
            if (ov < best[r] || (ov == best[r] && oi < bidx[r])) {
                best[r] = ov; bidx[r] = oi;
            }
        }
    }
    if ((lane & 15) == 0) {
#pragma unroll
        for (int r = 0; r < 8; ++r) idx[m0 + r] = bidx[r];
    }
}

// quant / straight-through output + commitment-loss sum
__global__ void vq_gather_kernel(const float* __restrict__ z,
                                 const float* __restrict__ emb,
                                 const int* __restrict__ idx,
                                 float* __restrict__ out_st,
                                 float* __restrict__ loss_acc) {
    int gid = blockIdx.x * 256 + threadIdx.x;     // 8388608
    int b = gid >> 20, c = (gid >> 12) & 255, thw = gid & 4095;
    int token = (b << 12) | thw;
    int id = idx[token];
    float q = emb[id * KDIM + c];
    out_st[gid] = q;
    float diff = z[gid] - q;
    __shared__ float sm[256];
    sm[threadIdx.x] = diff * diff;
    __syncthreads();
    for (int s = 128; s > 0; s >>= 1) {
        if (threadIdx.x < (unsigned)s) sm[threadIdx.x] += sm[threadIdx.x + s];
        __syncthreads();
    }
    if (threadIdx.x == 0) atomicAdd(loss_acc, sm[0]);
}

// one wave per token: counts + encode_sum scatter, encodings output
__global__ void vq_scatter_kernel(const float* __restrict__ z,
                                  const int* __restrict__ idx,
                                  float* __restrict__ counts,
                                  float* __restrict__ esum,
                                  float* __restrict__ out_enc) {
    int lane  = threadIdx.x & 31;
    int token = blockIdx.x * 8 + (threadIdx.x >> 5);   // 32768 waves
    int id = idx[token];
    int b = token >> 12, thw = token & 4095;
    if (lane == 0) {
        atomicAdd(&counts[id], 1.f);
        out_enc[token] = (float)id;
    }
    for (int c = lane; c < KDIM; c += 32) {
        float x = z[((size_t)(b * KDIM + c)) * THW + thw];
        atomicAdd(&esum[id * KDIM + c], x);
    }
}

// single block: n = sum(new_N), perplexity, commitment loss scalars
__global__ void vq_stats_kernel(const float* __restrict__ counts,
                                const float* __restrict__ N,
                                float* __restrict__ scal,
                                float* __restrict__ out_loss,
                                float* __restrict__ out_perp) {
    __shared__ float s_n[256], s_p[256];
    float nsum = 0.f, plog = 0.f;
    for (int i = threadIdx.x; i < NCODE; i += 256) {
        float cnt = counts[i];
        nsum += N[i] * 0.99f + 0.01f * cnt;
        float ap = cnt * (1.f / NTOK);
        plog += ap * logf(ap + 1e-10f);
    }
    s_n[threadIdx.x] = nsum; s_p[threadIdx.x] = plog;
    __syncthreads();
    for (int s = 128; s > 0; s >>= 1) {
        if (threadIdx.x < (unsigned)s) {
            s_n[threadIdx.x] += s_n[threadIdx.x + s];
            s_p[threadIdx.x] += s_p[threadIdx.x + s];
        }
        __syncthreads();
    }
    if (threadIdx.x == 0) {
        scal[1] = s_n[0];
        *out_perp = expf(-s_p[0]);
        *out_loss = scal[0] * (0.25f / (float)ZTOT);
    }
}

__global__ void vq_final_kernel(const float* __restrict__ counts,
                                const float* __restrict__ N,
                                const float* __restrict__ z_avg,
                                const float* __restrict__ k_rand,
                                const float* __restrict__ esum,
                                const float* __restrict__ scal,
                                float* __restrict__ out_newe,
                                float* __restrict__ out_newn,
                                float* __restrict__ out_newz) {
    int gid = blockIdx.x * 256 + threadIdx.x;     // 524288
    int code = gid >> 8, c = gid & 255;
    float cnt  = counts[code];
    float newN = N[code] * 0.99f + 0.01f * cnt;
    float nsum = scal[1];
    float w = (newN + 1e-7f) / (nsum + (float)NCODE * 1e-7f) * nsum;
    float nz = z_avg[gid] * 0.99f + 0.01f * esum[gid];
    float ne = nz / w;
    out_newe[gid] = (newN >= 1.f) ? ne : k_rand[gid];
    out_newz[gid] = nz;
    if (c == 0) out_newn[code] = newN;
}

extern "C" void kernel_launch(void* const* d_in, const int* in_sizes, int n_in,
                              void* d_out, int out_size, void* d_ws, size_t ws_size,
                              hipStream_t stream) {
    const float* z      = (const float*)d_in[0];
    const float* emb    = (const float*)d_in[1];
    const float* N      = (const float*)d_in[2];
    const float* z_avg  = (const float*)d_in[3];
    const float* k_rand = (const float*)d_in[4];
    float* o = (float*)d_out;

    // workspace layout (floats)
    float* wsf        = (float*)d_ws;
    float* counts     = wsf;                         // 2048
    float* encode_sum = counts + 2048;               // 524288
    float* x_norm     = encode_sum + 524288;         // 32768
    float* e_norm     = x_norm + 32768;              // 2048
    float* scal       = e_norm + 2048;               // 32  ([0]=loss_sum,[1]=nsum)
    int*   idx        = (int*)(scal + 32);           // 32768
    __bf16* A_hi = (__bf16*)(idx + 32768);           // 8388608 bf16
    __bf16* A_lo = A_hi + (size_t)NTOK * KDIM;
    __bf16* B_hi = A_lo + (size_t)NTOK * KDIM;       // 524288 bf16
    __bf16* B_lo = B_hi + (size_t)NCODE * KDIM;

    const int ZN = 2048 + 524288 + 32768 + 2048 + 32;   // accumulators to zero
    vq_zero_kernel<<<(ZN + 255) / 256, 256, 0, stream>>>(wsf, ZN);
    vq_prep_a_kernel<<<2048, 256, 0, stream>>>(z, A_hi, A_lo, x_norm);
    vq_prep_b_kernel<<<128, 256, 0, stream>>>(emb, B_hi, B_lo, e_norm);
    vq_argmin_kernel<<<256, 256, 0, stream>>>(A_hi, A_lo, B_hi, B_lo,
                                              x_norm, e_norm, idx);
    vq_gather_kernel<<<ZTOT / 256, 256, 0, stream>>>(z, emb, idx, o + O_ST, scal);
    vq_scatter_kernel<<<NTOK / 8, 256, 0, stream>>>(z, idx, counts, encode_sum,
                                                    o + O_ENC);
    vq_stats_kernel<<<1, 256, 0, stream>>>(counts, N, scal, o + O_LOSS, o + O_PERP);
    vq_final_kernel<<<(NCODE * KDIM) / 256, 256, 0, stream>>>(
        counts, N, z_avg, k_rand, encode_sum, scal,
        o + O_NEWE, o + O_NEWN, o + O_NEWZ);
}
